// MultiHeadAttention_83391085019519
// MI455X (gfx1250) — compile-verified, matrix operand order checked by reading, not run on hardware
//
#include <hip/hip_runtime.h>
#include <hip/hip_bf16.h>

// ---------------------------------------------------------------------------
// Causal MHA forward for B=2, T=2048, C=1024, H=16, D=64 on gfx1250.
// All matmuls use v_wmma_f32_16x16x32_bf16 (bf16 operands, f32 accum).
// Attention uses flash-attention streaming softmax (no TxT score matrix).
// All WMMA fragment builds are arranged to be two contiguous 16-byte LDS
// loads (ds_load_b128) per fragment: weights are pre-transposed at the
// f32->bf16 conversion step, and V is stored transposed (B,H,D,T).
// ---------------------------------------------------------------------------

typedef __attribute__((ext_vector_type(16))) __bf16 v16bf;
typedef __attribute__((ext_vector_type(8)))  float  v8f;

union V16 {
    v16bf v;
    unsigned short u[16];
};

#define T_SEQ 2048
#define HEADS 16
#define DHEAD 64
#define CDIM  1024
#define MROWS 4096   // B * T

static __device__ inline unsigned short f2bf(float f) {
    union { float f; unsigned u; } x; x.f = f;
    unsigned r = x.u + 0x7FFFu + ((x.u >> 16) & 1u);  // round-to-nearest-even
    return (unsigned short)(r >> 16);
}

static __device__ inline v8f vzero8() {
    v8f z;
#pragma unroll
    for (int i = 0; i < 8; ++i) z[i] = 0.0f;
    return z;
}

// Build a 16x32 bf16 A/B fragment from one LDS row: two contiguous 16B loads.
// Element mapping: u[2v+p] = row[base + (v>>2)*16 + half*8 + (v&3)*2 + p]
//  -> u[0..7] = row[base + half*8 .. +7], u[8..15] = row[base+16+half*8 .. +7]
static __device__ inline void frag_from_row(V16& f, const unsigned short* row,
                                            int base, int half) {
    *(uint4*)&f.u[0] = *(const uint4*)(row + base + half * 8);
    *(uint4*)&f.u[8] = *(const uint4*)(row + base + 16 + half * 8);
}

// ---------------------------------------------------------------------------
// f32 -> bf16 elementwise conversion
// ---------------------------------------------------------------------------
__global__ void cvt_f32_bf16(const float* __restrict__ in,
                             unsigned short* __restrict__ out, int n) {
    int i = blockIdx.x * blockDim.x + threadIdx.x;
    if (i < n) out[i] = f2bf(in[i]);
}

// f32 (K=1024 x N=1024) -> bf16 transposed (N x K)
__global__ void cvt_f32_bf16_t(const float* __restrict__ in,
                               unsigned short* __restrict__ out) {
    int i = blockIdx.x * blockDim.x + threadIdx.x;  // i = k*1024 + n
    int k = i >> 10, n = i & (CDIM - 1);
    out[(size_t)n * CDIM + k] = f2bf(in[i]);
}

// ---------------------------------------------------------------------------
// Tiled bf16 WMMA GEMM:  Y(4096 x 1024) = X(4096x1024) @ W(1024x1024)
// W is provided TRANSPOSED (Wt: N x K, row-major).
// Block = 128 threads (4 waves); block tile 128(M) x 64(N); wave tile 32x64.
// mode 0: write bf16 (B,H,T,D)                         (Q projection)
// mode 1: write f32 (B,H,T,D) AND bf16 (B,H,T,D)       (K projection)
// mode 3: write f32 (B,H,T,D) AND bf16 (B,H,D,T) [V^T] (V projection)
// mode 2: write f32 (B,T,C) with bias                  (output projection)
// ---------------------------------------------------------------------------
__global__ __launch_bounds__(128)
void gemm_bf16_wmma(const unsigned short* __restrict__ X,
                    const unsigned short* __restrict__ Wt,
                    float* __restrict__ out_f32,
                    unsigned short* __restrict__ out_bf,
                    const float* __restrict__ bias,
                    int mode) {
    __shared__ __align__(16) unsigned short xs[128][32];   // 8 KB
    __shared__ __align__(16) unsigned short wl_t[64][32];  // 4 KB (cols x K)

    const int tid  = threadIdx.x;
    const int lane = tid & 31;
    const int wv   = tid >> 5;          // wave id 0..3
    const int nlo  = lane & 15;
    const int half = lane >> 4;
    const int r0   = blockIdx.x * 128;
    const int c0   = blockIdx.y * 64;

    v8f acc[2][4];
#pragma unroll
    for (int rt = 0; rt < 2; ++rt)
#pragma unroll
        for (int nt = 0; nt < 4; ++nt) acc[rt][nt] = vzero8();

    for (int kc = 0; kc < CDIM / 32; ++kc) {
        const int k0 = kc * 32;
        // Stage X tile (128 rows x 32 K): 512 uint4 across 128 threads
#pragma unroll
        for (int j = 0; j < 4; ++j) {
            int idx = tid + 128 * j;
            int row = idx >> 2, qu = idx & 3;
            ((uint4*)&xs[0][0])[idx] =
                *(const uint4*)(X + (size_t)(r0 + row) * CDIM + k0 + qu * 8);
        }
        // Stage Wt tile (64 cols x 32 K): 256 uint4
#pragma unroll
        for (int j = 0; j < 2; ++j) {
            int idx = tid + 128 * j;
            int row = idx >> 2, qu = idx & 3;
            ((uint4*)&wl_t[0][0])[idx] =
                *(const uint4*)(Wt + (size_t)(c0 + row) * CDIM + k0 + qu * 8);
        }
        __syncthreads();

        V16 a[2];
#pragma unroll
        for (int rt = 0; rt < 2; ++rt)
            frag_from_row(a[rt], &xs[wv * 32 + rt * 16 + nlo][0], 0, half);
#pragma unroll
        for (int nt = 0; nt < 4; ++nt) {
            V16 bb;
            frag_from_row(bb, &wl_t[nt * 16 + nlo][0], 0, half);
#pragma unroll
            for (int rt = 0; rt < 2; ++rt)
                acc[rt][nt] = __builtin_amdgcn_wmma_f32_16x16x32_bf16(
                    false, a[rt].v, false, bb.v, (short)0, acc[rt][nt], false, false);
        }
        __syncthreads();
    }

    // Epilogue
#pragma unroll
    for (int rt = 0; rt < 2; ++rt) {
#pragma unroll
        for (int nt = 0; nt < 4; ++nt) {
#pragma unroll
            for (int v = 0; v < 8; ++v) {
                int r = r0 + wv * 32 + rt * 16 + v + 8 * half;
                int c = c0 + nt * 16 + nlo;
                float val = acc[rt][nt][v];
                if (mode == 2) {
                    out_f32[(size_t)r * CDIM + c] = val + bias[c];
                } else {
                    int b = r >> 11, t = r & (T_SEQ - 1);
                    int h = c >> 6,  d = c & (DHEAD - 1);
                    size_t idx = (((size_t)(b * HEADS + h)) * T_SEQ + t) * DHEAD + d;
                    if (mode == 3) {
                        out_f32[idx] = val;
                        size_t idxT = (((size_t)(b * HEADS + h)) * DHEAD + d) * T_SEQ + t;
                        out_bf[idxT] = f2bf(val);
                    } else {
                        out_bf[idx] = f2bf(val);
                        if (mode == 1) out_f32[idx] = val;
                    }
                }
            }
        }
    }
}

// ---------------------------------------------------------------------------
// Flash attention: one wave per (b, h, 16-query-row tile).
// Q/K in (B,H,T,D) bf16; V provided transposed (B,H,D,T) bf16 so that
// P.V B-fragments are contiguous LDS reads.
// ---------------------------------------------------------------------------
__global__ __launch_bounds__(32)
void flash_attn_wmma(const unsigned short* __restrict__ qbf,
                     const unsigned short* __restrict__ kbf,
                     const unsigned short* __restrict__ vbfT,
                     unsigned short* __restrict__ attbf) {
    __shared__ __align__(16) unsigned short k_lds[32][DHEAD];  // keys x d, 4 KB
    __shared__ __align__(16) unsigned short vt_lds[DHEAD][32]; // d x keys, 4 KB
    __shared__ __align__(16) unsigned short q_lds[16][DHEAD];  // 2 KB
    __shared__ __align__(16) unsigned short p_lds[16][32];     // 1 KB

    const int lane = threadIdx.x;
    const int n    = lane & 15;
    const int half = lane >> 4;
    const int blk  = blockIdx.x;              // b*H*(T/16) + h*(T/16) + qt
    const int qt   = blk & 127;
    const int h    = (blk >> 7) & (HEADS - 1);
    const int b    = blk >> 11;

    const size_t bh_off = ((size_t)(b * HEADS + h)) * T_SEQ * DHEAD;

    // Load 16x64 Q tile (128 uint4, 4 per lane)
    {
        const uint4* s4 = (const uint4*)(qbf + bh_off + (size_t)qt * 16 * DHEAD);
        uint4* dst = (uint4*)&q_lds[0][0];
#pragma unroll
        for (int i = 0; i < 4; ++i) dst[lane + 32 * i] = s4[lane + 32 * i];
    }
    __syncthreads();

    // Q in A-layout registers: two K=32 chunks covering D=64
    V16 a_q[2];
#pragma unroll
    for (int c = 0; c < 2; ++c)
        frag_from_row(a_q[c], &q_lds[n][0], c * 32, half);

    v8f acc_o[4];
#pragma unroll
    for (int i = 0; i < 4; ++i) acc_o[i] = vzero8();
    float m_run[8], l_run[8];
#pragma unroll
    for (int v = 0; v < 8; ++v) { m_run[v] = -1e30f; l_run[v] = 0.0f; }

    const int nst = (qt * 16 + 47) / 32;  // ceil((last_row+1)/32), causal

    for (int st = 0; st < nst; ++st) {
        // Stage K tile (32 keys x 64 d) and V^T tile (64 d x 32 keys)
        {
            const uint4* ks4 = (const uint4*)(kbf + bh_off + (size_t)st * 32 * DHEAD);
            uint4* kd = (uint4*)&k_lds[0][0];
#pragma unroll
            for (int i = 0; i < 8; ++i) kd[lane + 32 * i] = ks4[lane + 32 * i];

            uint4* vd = (uint4*)&vt_lds[0][0];
#pragma unroll
            for (int i = 0; i < 8; ++i) {
                int idx = lane + 32 * i;          // 256 uint4
                int row = idx >> 2, qu = idx & 3; // row = d (0..63)
                vd[idx] = *(const uint4*)(vbfT + bh_off + (size_t)row * T_SEQ +
                                          st * 32 + qu * 8);
            }
        }
        __syncthreads();

        // S = Q K^T : two 16x16 tiles (keys 0-15, 16-31), K-dim 64 in 2 chunks
        v8f acc_s[2];
#pragma unroll
        for (int t = 0; t < 2; ++t) {
            acc_s[t] = vzero8();
#pragma unroll
            for (int c = 0; c < 2; ++c) {
                V16 bk;
                frag_from_row(bk, &k_lds[t * 16 + n][0], c * 32, half);
                acc_s[t] = __builtin_amdgcn_wmma_f32_16x16x32_bf16(
                    false, a_q[c].v, false, bk.v, (short)0, acc_s[t], false, false);
            }
        }

        // Online softmax (rows map to (v, half); row reduce = xor 1,2,4,8)
        float p0[8], p1[8], rowmax[8];
#pragma unroll
        for (int v = 0; v < 8; ++v) {
            int row_g = qt * 16 + v + 8 * half;
            float s0 = acc_s[0][v] * 0.125f;   // 1/sqrt(64)
            float s1 = acc_s[1][v] * 0.125f;
            if (st * 32 + n      > row_g) s0 = -1e30f;
            if (st * 32 + 16 + n > row_g) s1 = -1e30f;
            p0[v] = s0; p1[v] = s1;
            rowmax[v] = fmaxf(s0, s1);
        }
#pragma unroll
        for (int mask = 1; mask < 16; mask <<= 1)
#pragma unroll
            for (int v = 0; v < 8; ++v)
                rowmax[v] = fmaxf(rowmax[v], __shfl_xor(rowmax[v], mask, 32));

        float alpha[8], rowsum[8];
#pragma unroll
        for (int v = 0; v < 8; ++v) {
            float mn = fmaxf(m_run[v], rowmax[v]);
            alpha[v] = __expf(m_run[v] - mn);
            m_run[v] = mn;
            p0[v] = __expf(p0[v] - mn);
            p1[v] = __expf(p1[v] - mn);
            rowsum[v] = p0[v] + p1[v];
        }
#pragma unroll
        for (int mask = 1; mask < 16; mask <<= 1)
#pragma unroll
            for (int v = 0; v < 8; ++v)
                rowsum[v] += __shfl_xor(rowsum[v], mask, 32);
#pragma unroll
        for (int v = 0; v < 8; ++v) l_run[v] = l_run[v] * alpha[v] + rowsum[v];
#pragma unroll
        for (int nt = 0; nt < 4; ++nt)
#pragma unroll
            for (int v = 0; v < 8; ++v) acc_o[nt][v] *= alpha[v];

        // P (D-layout) -> LDS -> A-layout bf16
#pragma unroll
        for (int v = 0; v < 8; ++v) {
            int m = v + 8 * half;
            p_lds[m][n]      = f2bf(p0[v]);
            p_lds[m][16 + n] = f2bf(p1[v]);
        }
        __syncthreads();

        V16 a_p;
        frag_from_row(a_p, &p_lds[n][0], 0, half);

        // O += P V : 4 column tiles of D=64; B fragment rows from V^T LDS
#pragma unroll
        for (int nt = 0; nt < 4; ++nt) {
            V16 bv;
            frag_from_row(bv, &vt_lds[nt * 16 + n][0], 0, half);
            acc_o[nt] = __builtin_amdgcn_wmma_f32_16x16x32_bf16(
                false, a_p.v, false, bv.v, (short)0, acc_o[nt], false, false);
        }
        __syncthreads();
    }

    // Normalize and write att output as bf16 in (B,T,C) layout
#pragma unroll
    for (int nt = 0; nt < 4; ++nt)
#pragma unroll
        for (int v = 0; v < 8; ++v) {
            int m = v + 8 * half;
            int t = qt * 16 + m;
            int c = h * DHEAD + nt * 16 + n;
            float val = acc_o[nt][v] / l_run[v];
            attbf[((size_t)(b * T_SEQ + t)) * CDIM + c] = f2bf(val);
        }
}

// ---------------------------------------------------------------------------
// Host-side launch
// ---------------------------------------------------------------------------
extern "C" void kernel_launch(void* const* d_in, const int* in_sizes, int n_in,
                              void* d_out, int out_size, void* d_ws, size_t ws_size,
                              hipStream_t stream) {
    const float* x  = (const float*)d_in[0];
    const float* Wq = (const float*)d_in[1];
    const float* Wk = (const float*)d_in[2];
    const float* Wv = (const float*)d_in[3];
    const float* Wo = (const float*)d_in[4];
    const float* bo = (const float*)d_in[5];

    float* out  = (float*)d_out;                          // (B,T,C) f32
    float* kout = out  + (size_t)MROWS * CDIM;            // (B,H,T,D) f32
    float* vout = kout + (size_t)MROWS * CDIM;            // (B,H,T,D) f32

    // Workspace layout (bytes)
    char* ws = (char*)d_ws;
    unsigned short* xbf   = (unsigned short*)(ws);                    // 8 MB
    unsigned short* wqbf  = (unsigned short*)(ws + (8u  << 20));      // 2 MB (transposed)
    unsigned short* wkbf  = (unsigned short*)(ws + (10u << 20));      // 2 MB (transposed)
    unsigned short* wvbf  = (unsigned short*)(ws + (12u << 20));      // 2 MB (transposed)
    unsigned short* wobf  = (unsigned short*)(ws + (14u << 20));      // 2 MB (transposed)
    unsigned short* qbf   = (unsigned short*)(ws + (16u << 20));      // 8 MB (B,H,T,D)
    unsigned short* kbf   = (unsigned short*)(ws + (24u << 20));      // 8 MB (B,H,T,D)
    unsigned short* vbfT  = (unsigned short*)(ws + (32u << 20));      // 8 MB (B,H,D,T)
    unsigned short* attbf = (unsigned short*)(ws + (40u << 20));      // 8 MB (B,T,C)

    const int NX = MROWS * CDIM;   // 4,194,304
    const int NW = CDIM * CDIM;    // 1,048,576

    cvt_f32_bf16  <<<(NX + 255) / 256, 256, 0, stream>>>(x,  xbf, NX);
    cvt_f32_bf16_t<<<(NW + 255) / 256, 256, 0, stream>>>(Wq, wqbf);
    cvt_f32_bf16_t<<<(NW + 255) / 256, 256, 0, stream>>>(Wk, wkbf);
    cvt_f32_bf16_t<<<(NW + 255) / 256, 256, 0, stream>>>(Wv, wvbf);
    cvt_f32_bf16_t<<<(NW + 255) / 256, 256, 0, stream>>>(Wo, wobf);

    dim3 gg(MROWS / 128, CDIM / 64), gb(128);
    gemm_bf16_wmma<<<gg, gb, 0, stream>>>(xbf, wqbf, nullptr, qbf,  nullptr, 0);
    gemm_bf16_wmma<<<gg, gb, 0, stream>>>(xbf, wkbf, kout,    kbf,  nullptr, 1);
    gemm_bf16_wmma<<<gg, gb, 0, stream>>>(xbf, wvbf, vout,    vbfT, nullptr, 3);

    flash_attn_wmma<<<2 * HEADS * (T_SEQ / 16), 32, 0, stream>>>(qbf, kbf, vbfT, attbf);

    gemm_bf16_wmma<<<gg, gb, 0, stream>>>(attbf, wobf, out, nullptr, bo, 2);
}
